// PhasorMultiDense_17798344475244
// MI455X (gfx1250) — matrix operand order repaired and checked by reading, not compiled, not run on hardware
//
#include <hip/hip_runtime.h>
#include <math.h>

// ---------------------------------------------------------------------------
// PhasorMultiDense for MI455X (gfx1250, wave32, WMMA + TDM)
//
// Shapes: B=16, T=2048, D=1024, H=16, O=64;  M = B*T = 32768
// Pass 1a: A_re/A_im[m][d] f16 = mask(|x|>=0.5) * cospi/sinpi(x)   (64MB each)
// Pass 1b: Wt[h][o][d] f16 = transpose+convert of w[h][d][o]       (2MB)
// Pass 2 : per-head GEMM via v_wmma_f32_16x16x32_f16, f32 accum.
//          W tiles double-buffered in LDS via TENSOR_LOAD_TO_LDS (TDM),
//          with hardware LDS padding for bank-conflict-free ds_load_b128.
//          Epilogue: atan2(Im,Re)/pi with re-mask, scatter to (B,H,T,O).
// Workspace layout: [0,64MB) A_re | [64,128MB) A_im | [128,130MB) Wt
// ---------------------------------------------------------------------------

typedef _Float16 v16h __attribute__((ext_vector_type(16)));
typedef _Float16 v8h  __attribute__((ext_vector_type(8)));
typedef float    v8f  __attribute__((ext_vector_type(8)));
typedef unsigned int u32x4 __attribute__((ext_vector_type(4)));
typedef int          i32x4 __attribute__((ext_vector_type(4)));
typedef int          i32x8 __attribute__((ext_vector_type(8)));

#define BB 16
#define TT 2048
#define DD 1024
#define HH 16
#define OO 64
#define MM (BB * TT)          // 32768

// ---------------------------------------------------------------------------
// Pass 1a: phase encode. 8 elements / thread, b128 in, b128 out x2.
// ---------------------------------------------------------------------------
__global__ __launch_bounds__(256) void phase_encode_kernel(
    const float* __restrict__ x, _Float16* __restrict__ are,
    _Float16* __restrict__ aim) {
  const long tid = (long)blockIdx.x * 256 + threadIdx.x;   // [0, M*D/8)
  const float4* xv = (const float4*)x;
  float4 v0 = xv[tid * 2 + 0];
  float4 v1 = xv[tid * 2 + 1];
  float vv[8] = {v0.x, v0.y, v0.z, v0.w, v1.x, v1.y, v1.z, v1.w};
  v8h r16, i16;
#pragma unroll
  for (int i = 0; i < 8; ++i) {
    float v = vv[i];
    bool m = __builtin_fabsf(v) >= 0.5f;          // MASK_ANGLE = 0.5
    float c = m ? cospif(v) : 0.0f;               // exp(i*pi*x) masked
    float s = m ? sinpif(v) : 0.0f;
    r16[i] = (_Float16)c;
    i16[i] = (_Float16)s;
  }
  ((v8h*)are)[tid] = r16;
  ((v8h*)aim)[tid] = i16;
}

// ---------------------------------------------------------------------------
// Pass 1b: Wt[h][o][d] = (f16) w[h][d][o]. 1M elements; writes coalesced.
// ---------------------------------------------------------------------------
__global__ __launch_bounds__(256) void convert_w_kernel(
    const float* __restrict__ w, _Float16* __restrict__ wt) {
  const int idx = blockIdx.x * 256 + threadIdx.x;  // [0, H*O*D)
  const int h = idx >> 16;                          // O*D = 65536 per head
  const int r = idx & 65535;
  const int o = r >> 10;
  const int d = r & 1023;
  wt[(((long)(h * OO + o)) << 10) + d] =
      (_Float16)w[((long)(h * DD + d)) * OO + o];
}

// ---------------------------------------------------------------------------
// TDM: load one W tile (64 rows x 128 halves, row stride 1024 halves) into
// LDS with hardware padding of 4 DWORDs every 64 DWORDs -> padded row stride
// of 136 halves (272B -> bank stride 68 = 64+4: conflict-free b128 reads).
// D# packing per CDNA5 ISA ch.8 (group0 128b, group1 256b).
// ---------------------------------------------------------------------------
#define KCHUNK 128
#define WROW   136   // 128 + 8 pad halves

__device__ __forceinline__ unsigned lds_offset_of(void* p) {
  // generic -> LDS addrspace cast yields the 32-bit LDS byte offset
  return (unsigned)(size_t)(__attribute__((address_space(3))) void*)p;
}

__device__ __forceinline__ void tdm_load_wtile(unsigned lds_off,
                                               const _Float16* gsrc) {
  const unsigned long long ga = (unsigned long long)(size_t)gsrc;
  u32x4 g0;
  g0.x = 1u;                                            // count=1 (valid D#)
  g0.y = lds_off;                                       // lds_addr (bytes)
  g0.z = (unsigned)ga;                                  // global_addr[31:0]
  g0.w = (unsigned)((ga >> 32) & 0x01FFFFFFull) | (2u << 30);  // addr|type=2
  i32x8 g1;
  g1[0] = (1 << 16)      // data_size = 1 -> 2 bytes/elem
        | (1 << 20)      // pad_enable
        | (5 << 22)      // pad_interval: 2^(5+1)=64 DWORDs (= one 128h row)
        | (3 << 25);     // pad_amount: 3+1=4 DWORDs (= 8 halves)
  g1[1] = (int)((DD & 0xFFFF) << 16);                   // tensor_dim0 lo16
  g1[2] = (int)((DD >> 16) | (OO << 16));               // dim0 hi | dim1 lo
  g1[3] = (int)((OO >> 16) | (KCHUNK << 16));           // dim1 hi | tile_dim0
  g1[4] = OO;                                           // tile_dim1 | tile_dim2=0
  g1[5] = DD;                                           // tensor_dim0_stride lo32
  g1[6] = 0;                                            // stride0 hi | stride1 lo
  g1[7] = 0;
  const i32x4 z4 = {0, 0, 0, 0};
#if defined(__clang_major__) && __clang_major__ >= 23
  const i32x8 z8 = {0, 0, 0, 0, 0, 0, 0, 0};
  __builtin_amdgcn_tensor_load_to_lds(g0, g1, z4, z4, z8, 0);
#else
  __builtin_amdgcn_tensor_load_to_lds(g0, g1, z4, z4, 0);
#endif
}

// ---------------------------------------------------------------------------
// Pass 2: GEMM.  grid = (H, M/128), block = 256 (8 waves).
// Wave w: rows [mb*128 + w*16, +16), cols = one head's 64 outputs.
// ---------------------------------------------------------------------------
__global__ __launch_bounds__(256) void phasor_gemm_kernel(
    const _Float16* __restrict__ are, const _Float16* __restrict__ aim,
    const _Float16* __restrict__ wt, const float* __restrict__ bzre,
    const float* __restrict__ bzim, float* __restrict__ out) {
  __shared__ __align__(16) _Float16 Ws[2][OO][WROW];    // 2 x 17KB

  const int h    = blockIdx.x;           // fastest grid dim: heads of the same
  const int mb   = blockIdx.y;           // M-chunk co-scheduled -> A hot in L2
  const int tid  = threadIdx.x;
  const int wave = tid >> 5;
  const int lane = tid & 31;
  const int lhi  = lane >> 4;            // 0: lanes 0-15, 1: lanes 16-31
  const int lmod = lane & 15;

  const int  m0   = mb * 128 + wave * 16;
  const long arow = (long)(m0 + lmod) * DD;        // this lane's A row
  const _Float16* wth = wt + (((long)h * OO) << 10);

  // Scalar (wave-uniform) predicate: TDM issues regardless of EXEC, so the
  // branch MUST be a scalar branch, not EXEC masking -> readfirstlane.
  const bool issuer = (__builtin_amdgcn_readfirstlane(wave) == 0);
  const unsigned lds0 = lds_offset_of(&Ws[0][0][0]);
  const unsigned lds1 = lds_offset_of(&Ws[1][0][0]);

  const v8f vzero = {0.f, 0.f, 0.f, 0.f, 0.f, 0.f, 0.f, 0.f};
  v8f accre[4], accim[4];
#pragma unroll
  for (int nt = 0; nt < 4; ++nt) { accre[nt] = vzero; accim[nt] = vzero; }

  if (issuer) tdm_load_wtile(lds0, wth);           // prefetch chunk 0

  for (int c = 0; c < DD / KCHUNK; ++c) {          // 8 chunks of K=128
    if (issuer) {
      if (c < DD / KCHUNK - 1) {
        // overlap: issue chunk c+1 into the other buffer, then wait for
        // the *older* (chunk c) transfer only.  TENSORcnt is in-order.
        tdm_load_wtile((c & 1) ? lds0 : lds1,
                       wth + (long)(c + 1) * KCHUNK);
        __builtin_amdgcn_s_wait_tensorcnt(1);
      } else {
        __builtin_amdgcn_s_wait_tensorcnt(0);
      }
    }
    __syncthreads();                               // chunk c visible to all

    const int kc = c * KCHUNK;
    const _Float16(*Wb)[WROW] = Ws[c & 1];
#pragma unroll
    for (int k0 = 0; k0 < KCHUNK; k0 += 32) {
      // A fragments, register-direct from global (ISA 16-bit A layout):
      // lanes 0-15 : VGPR0-3 = K 0-7,  VGPR4-7 = K 16-23
      // lanes 16-31: VGPR0-3 = K 8-15, VGPR4-7 = K 24-31
      const _Float16* pr = are + arow + kc + k0 + lhi * 8;
      const _Float16* pi = aim + arow + kc + k0 + lhi * 8;
      v8h arlo = *(const v8h*)(pr);
      v8h arhi = *(const v8h*)(pr + 16);
      v8h ailo = *(const v8h*)(pi);
      v8h aihi = *(const v8h*)(pi + 16);
      v16h afr = __builtin_shufflevector(arlo, arhi, 0, 1, 2, 3, 4, 5, 6, 7,
                                         8, 9, 10, 11, 12, 13, 14, 15);
      v16h afi = __builtin_shufflevector(ailo, aihi, 0, 1, 2, 3, 4, 5, 6, 7,
                                         8, 9, 10, 11, 12, 13, 14, 15);
#pragma unroll
      for (int nt = 0; nt < 4; ++nt) {
        // B fragment (same K order as A): lane = N, lanes 0-15 hold
        // K 0-7,16-23, lanes 16-31 hold K 8-15,24-31.
        const _Float16* pb = &Wb[nt * 16 + lmod][k0 + lhi * 8];
        v8h blo = *(const v8h*)(pb);
        v8h bhi = *(const v8h*)(pb + 16);
        v16h bf = __builtin_shufflevector(blo, bhi, 0, 1, 2, 3, 4, 5, 6, 7,
                                          8, 9, 10, 11, 12, 13, 14, 15);
        accre[nt] = __builtin_amdgcn_wmma_f32_16x16x32_f16(
            false, afr, false, bf, (short)0, accre[nt], false, false);
        accim[nt] = __builtin_amdgcn_wmma_f32_16x16x32_f16(
            false, afi, false, bf, (short)0, accim[nt], false, false);
      }
    }
    __syncthreads();   // all reads of this buffer done before TDM reuses it
  }

  // Epilogue: +bias, angle/pi, re-mask, scatter to (B,H,T,O).
  const float INV_PI = 0.318309886183790671538f;
#pragma unroll
  for (int nt = 0; nt < 4; ++nt) {
    const int o = nt * 16 + lmod;
    const float br = bzre[h * OO + o];
    const float bi = bzim[h * OO + o];
#pragma unroll
    for (int j = 0; j < 8; ++j) {
      const int r = m0 + j + lhi * 8;        // C/D layout: VGPR j, lane half
      const float re = accre[nt][j] + br;
      const float im = accim[nt][j] + bi;
      float y = atan2f(im, re) * INV_PI;     // jnp.angle / pi
      y = (__builtin_fabsf(y) >= 0.5f) ? y : 0.0f;
      const int b = r >> 11;                 // r / T
      const int t = r & 2047;                // r % T
      out[(((size_t)(b * HH + h) * TT + t) << 6) + o] = y;
    }
  }
}

// ---------------------------------------------------------------------------
extern "C" void kernel_launch(void* const* d_in, const int* in_sizes, int n_in,
                              void* d_out, int out_size, void* d_ws,
                              size_t ws_size, hipStream_t stream) {
  (void)in_sizes; (void)n_in; (void)out_size; (void)ws_size;
  const float* x    = (const float*)d_in[0];   // (B,T,D) f32
  const float* w    = (const float*)d_in[1];   // (H,D,O) f32
  const float* bzre = (const float*)d_in[2];   // (H,O)   f32
  const float* bzim = (const float*)d_in[3];   // (H,O)   f32
  float* out = (float*)d_out;                  // (B,H,T,O) f32

  char* ws = (char*)d_ws;                      // needs 130 MB scratch
  _Float16* are = (_Float16*)(ws);
  _Float16* aim = (_Float16*)(ws + ((size_t)64 << 20));
  _Float16* wt  = (_Float16*)(ws + ((size_t)128 << 20));

  // Pass 1a: M*D/8 threads
  phase_encode_kernel<<<dim3((MM * DD) / (256 * 8)), 256, 0, stream>>>(
      x, are, aim);
  // Pass 1b: H*O*D threads
  convert_w_kernel<<<dim3((HH * OO * DD) / 256), 256, 0, stream>>>(w, wt);
  // Pass 2: heads fastest so all 16 heads of an M-chunk share A via L2
  phasor_gemm_kernel<<<dim3(HH, MM / 128), 256, 0, stream>>>(
      are, aim, wt, bzre, bzim, out);
}
